// MultiCell_layer_73641509257892
// MI455X (gfx1250) — matrix operand order repaired and checked by reading, not compiled, not run on hardware
//
#include <hip/hip_runtime.h>
#include <hip/hip_bf16.h>
#include <math.h>

// ---------------------------------------------------------------------------
// Problem constants (match reference)
// ---------------------------------------------------------------------------
#define BB 4096
#define II 1024
#define HH 1024
#define CC 16

typedef __attribute__((ext_vector_type(16))) __bf16 v16bf;
typedef __attribute__((ext_vector_type(8)))  __bf16 v8bf;
typedef __attribute__((ext_vector_type(8)))  float  v8f;

// ---------------------------------------------------------------------------
// WMMA fragment helpers (bf16 16x16x32, fp32 accumulate)
//
// A (16x32, MxK) per ISA layout: lane l holds row M = l&15; its 16 bf16 are
// K = kbase..kbase+7 and kbase+16..kbase+23 where kbase = k0 + 8*(l>>4).
// -> two contiguous 16-byte loads from a row-major bf16 matrix.
//
// B (32x16, KxN): lane l holds column N = l&15, same K order as A:
// kbase..kbase+7 and kbase+16..kbase+23 -> two 16-byte loads from an N-major (transposed) bf16 matrix.
//
// C/D (16x16 f32): VGPR r, lane l -> M = r + 8*(l>>4), N = l&15.
// ---------------------------------------------------------------------------
__device__ __forceinline__ v16bf load_a_frag(const __bf16* __restrict__ A,
                                             int lda, int row0, int k0, int lane) {
    int row = row0 + (lane & 15);
    int kb  = k0 + ((lane >> 4) << 3);
    const __bf16* p = A + (size_t)row * lda + kb;
    v8bf lo = *(const v8bf*)(p);
    v8bf hi = *(const v8bf*)(p + 16);
    return __builtin_shufflevector(lo, hi, 0, 1, 2, 3, 4, 5, 6, 7,
                                           8, 9, 10, 11, 12, 13, 14, 15);
}

__device__ __forceinline__ v16bf load_b_frag(const __bf16* __restrict__ Wt,
                                             int ldb, int n0, int k0, int lane) {
    int n  = n0 + (lane & 15);
    int kb = k0 + ((lane >> 4) << 3);
    const __bf16* p = Wt + (size_t)n * ldb + kb;
    v8bf lo = *(const v8bf*)(p);
    v8bf hi = *(const v8bf*)(p + 16);
    return __builtin_shufflevector(lo, hi, 0, 1, 2, 3, 4, 5, 6, 7,
                                           8, 9, 10, 11, 12, 13, 14, 15);
}

__device__ __forceinline__ v8f wmma_bf16(v16bf a, v16bf b, v8f c) {
    return __builtin_amdgcn_wmma_f32_16x16x32_bf16(false, a, false, b,
                                                   (short)0, c, false, false);
}

// 32x64 tile GEMM: acc[2][4] += A[m0:m0+32, :K] * Wt^T[n0:n0+64, :K]^T
__device__ __forceinline__ void gemm_tile_32x64(const __bf16* __restrict__ A,
                                                const __bf16* __restrict__ Wt,
                                                int K, int m0, int n0, int lane,
                                                v8f acc[2][4]) {
    for (int k0 = 0; k0 < K; k0 += 32) {
        v16bf a0 = load_a_frag(A, K, m0, k0, lane);
        v16bf a1 = load_a_frag(A, K, m0 + 16, k0, lane);
#pragma unroll
        for (int nt = 0; nt < 4; ++nt) {
            v16bf b = load_b_frag(Wt, K, n0 + nt * 16, k0, lane);
            acc[0][nt] = wmma_bf16(a0, b, acc[0][nt]);
            acc[1][nt] = wmma_bf16(a1, b, acc[1][nt]);
        }
    }
}

__device__ __forceinline__ float sigm(float x) { return 1.0f / (1.0f + expf(-x)); }

// ---------------------------------------------------------------------------
// Pack kernels
// ---------------------------------------------------------------------------
__global__ void cvt_f32_bf16_kernel(const float* __restrict__ src,
                                    __bf16* __restrict__ dst, size_t n) {
    size_t i = (size_t)blockIdx.x * blockDim.x + threadIdx.x;
    if (i < n) dst[i] = (__bf16)src[i];
}

__global__ void zero_f32_kernel(float* __restrict__ dst, size_t n) {
    size_t i = (size_t)blockIdx.x * blockDim.x + threadIdx.x;
    if (i < n) dst[i] = 0.0f;
}

// W[b][K][N] f32 (row-major) -> Wt[b][N][K] bf16 (N-major). K,N multiples of 32.
__global__ void transpose_pack_kernel(const float* __restrict__ src,
                                      __bf16* __restrict__ dst, int K, int N) {
    __shared__ float tile[32][33];
    const int b = blockIdx.z;
    const float* S = src + (size_t)b * K * N;
    __bf16* D = dst + (size_t)b * N * K;
    const int n0 = blockIdx.x * 32, k0 = blockIdx.y * 32;
    const int tx = threadIdx.x, ty = threadIdx.y; // 32 x 8
#pragma unroll
    for (int i = 0; i < 4; ++i) {
        int k = k0 + ty + i * 8;
        tile[ty + i * 8][tx] = S[(size_t)k * N + (n0 + tx)];
    }
    __syncthreads();
#pragma unroll
    for (int i = 0; i < 4; ++i) {
        int n = n0 + ty + i * 8;
        D[(size_t)n * K + (k0 + tx)] = (__bf16)tile[tx][ty + i * 8];
    }
}

// ---------------------------------------------------------------------------
// Cell-gate GEMM + fused gate mix.
// net[b,c,n] = X@Wcx[c] + Hs@Wch[c] + b[c];  n<H -> gate, n>=H -> input.
// cell_states = sig(gate)*tanh(inp) + (1-sig(gate))*pre
// ---------------------------------------------------------------------------
__global__ __launch_bounds__(32) void cellgate_kernel(
    const __bf16* __restrict__ Xb, const __bf16* __restrict__ Hb,
    const __bf16* __restrict__ WtCX, const __bf16* __restrict__ WtCH,
    const float* __restrict__ cell_x_b, const float* __restrict__ pre,
    float* __restrict__ out_cs, __bf16* __restrict__ ws_cs) {
    const int lane = threadIdx.x;
    const int n0 = blockIdx.x * 64;   // gate column block within [0, H)
    const int m0 = blockIdx.y * 32;
    const int c  = blockIdx.z;
    const __bf16* Bx = WtCX + (size_t)c * 2 * HH * II;
    const __bf16* Bh = WtCH + (size_t)c * 2 * HH * HH;

    v8f accG[2][4] = {};
    v8f accI[2][4] = {};
#pragma unroll 1
    for (int ph = 0; ph < 2; ++ph) {
        const __bf16* A  = ph ? Hb : Xb;
        const __bf16* Bp = ph ? Bh : Bx;
        for (int k0 = 0; k0 < II; k0 += 32) {
            v16bf a0 = load_a_frag(A, II, m0, k0, lane);
            v16bf a1 = load_a_frag(A, II, m0 + 16, k0, lane);
#pragma unroll
            for (int nt = 0; nt < 4; ++nt) {
                v16bf bg = load_b_frag(Bp, II, n0 + nt * 16, k0, lane);
                accG[0][nt] = wmma_bf16(a0, bg, accG[0][nt]);
                accG[1][nt] = wmma_bf16(a1, bg, accG[1][nt]);
                v16bf bi = load_b_frag(Bp, II, HH + n0 + nt * 16, k0, lane);
                accI[0][nt] = wmma_bf16(a0, bi, accI[0][nt]);
                accI[1][nt] = wmma_bf16(a1, bi, accI[1][nt]);
            }
        }
    }
    const int cl = lane & 15, rh = lane >> 4;
#pragma unroll
    for (int mt = 0; mt < 2; ++mt)
#pragma unroll
        for (int nt = 0; nt < 4; ++nt) {
            int col = n0 + nt * 16 + cl;
            float bg = cell_x_b[c * 2 * HH + col];
            float bi = cell_x_b[c * 2 * HH + HH + col];
#pragma unroll
            for (int r = 0; r < 8; ++r) {
                int row = m0 + mt * 16 + rh * 8 + r;
                float ig = sigm(accG[mt][nt][r] + bg);
                float ci = tanhf(accI[mt][nt][r] + bi);
                float pv = pre[(size_t)row * HH + col];
                float cs = ig * ci + (1.0f - ig) * pv;
                out_cs[((size_t)row * CC + c) * HH + col] = cs;          // [B,C,H]
                ws_cs[((size_t)c * BB + row) * HH + col] = (__bf16)cs;   // [C,B,H]
            }
        }
}

// ---------------------------------------------------------------------------
// Composition GEMM: comp = X@Wpx + Hs@Wph + b  (N = 3H), fused activations.
// ---------------------------------------------------------------------------
__global__ __launch_bounds__(32) void comp_kernel(
    const __bf16* __restrict__ Xb, const __bf16* __restrict__ Hb,
    const __bf16* __restrict__ WtPX, const __bf16* __restrict__ WtPH,
    const float* __restrict__ comp_x_b,
    float* __restrict__ ws_ig, float* __restrict__ ws_og,
    __bf16* __restrict__ ws_cc) {
    const int lane = threadIdx.x;
    const int n0 = blockIdx.x * 64;   // [0, 3H)
    const int m0 = blockIdx.y * 32;

    v8f acc[2][4] = {};
    gemm_tile_32x64(Xb, WtPX, II, m0, n0, lane, acc);
    gemm_tile_32x64(Hb, WtPH, HH, m0, n0, lane, acc);

    const int cl = lane & 15, rh = lane >> 4;
#pragma unroll
    for (int mt = 0; mt < 2; ++mt)
#pragma unroll
        for (int nt = 0; nt < 4; ++nt) {
            int col = n0 + nt * 16 + cl;
            float bb = comp_x_b[col];
#pragma unroll
            for (int r = 0; r < 8; ++r) {
                int row = m0 + mt * 16 + rh * 8 + r;
                float v = acc[mt][nt][r] + bb;
                if (col < HH) {
                    ws_ig[(size_t)row * HH + col] = sigm(v);
                } else if (col < 2 * HH) {
                    ws_og[(size_t)row * HH + (col - HH)] = sigm(v);
                } else {
                    ws_cc[(size_t)row * HH + (col - 2 * HH)] = (__bf16)tanhf(v);
                }
            }
        }
}

// ---------------------------------------------------------------------------
// u = cell_state_c @ attn_in_W   (f32 out, [B, 2H])
// ---------------------------------------------------------------------------
__global__ __launch_bounds__(32) void attn_u_kernel(
    const __bf16* __restrict__ CCb, const __bf16* __restrict__ WtAI,
    float* __restrict__ u) {
    const int lane = threadIdx.x;
    const int n0 = blockIdx.x * 64;   // [0, 2H)
    const int m0 = blockIdx.y * 32;
    v8f acc[2][4] = {};
    gemm_tile_32x64(CCb, WtAI, HH, m0, n0, lane, acc);
    const int cl = lane & 15, rh = lane >> 4;
#pragma unroll
    for (int mt = 0; mt < 2; ++mt)
#pragma unroll
        for (int nt = 0; nt < 4; ++nt) {
            int col = n0 + nt * 16 + cl;
#pragma unroll
            for (int r = 0; r < 8; ++r) {
                int row = m0 + mt * 16 + rh * 8 + r;
                u[(size_t)row * (2 * HH) + col] = acc[mt][nt][r];
            }
        }
}

// ---------------------------------------------------------------------------
// scores[b,c] += sum_n atten_v[n] * tanh(u[b,n] + (cs[c] @ attn_cell_W)[b,n])
// split over n (atomic accumulate); scores pre-zeroed.
// ---------------------------------------------------------------------------
__global__ __launch_bounds__(32) void attn_score_kernel(
    const __bf16* __restrict__ ws_cs, const __bf16* __restrict__ WtAC,
    const float* __restrict__ u, const float* __restrict__ av,
    float* __restrict__ scores) {
    const int lane = threadIdx.x;
    const int n0 = blockIdx.x * 64;
    const int m0 = blockIdx.y * 32;
    const int c  = blockIdx.z;
    const __bf16* A = ws_cs + (size_t)c * BB * HH;

    v8f acc[2][4] = {};
    gemm_tile_32x64(A, WtAC, HH, m0, n0, lane, acc);

    const int cl = lane & 15, rh = lane >> 4;
#pragma unroll
    for (int mt = 0; mt < 2; ++mt) {
        float part[8];
#pragma unroll
        for (int r = 0; r < 8; ++r) part[r] = 0.0f;
#pragma unroll
        for (int nt = 0; nt < 4; ++nt) {
            int col = n0 + nt * 16 + cl;
            float vv = av[col];
#pragma unroll
            for (int r = 0; r < 8; ++r) {
                int row = m0 + mt * 16 + rh * 8 + r;
                part[r] += vv * tanhf(u[(size_t)row * (2 * HH) + col] + acc[mt][nt][r]);
            }
        }
#pragma unroll
        for (int r = 0; r < 8; ++r) {
            float sv = part[r];
            sv += __shfl_xor(sv, 1);
            sv += __shfl_xor(sv, 2);
            sv += __shfl_xor(sv, 4);
            sv += __shfl_xor(sv, 8);
            if (cl == 0) {
                int row = m0 + mt * 16 + rh * 8 + r;
                atomicAdd(&scores[row * CC + c], sv);
            }
        }
    }
}

// ---------------------------------------------------------------------------
// Softmax over C, weighted cell sum, final gate mix.
// ---------------------------------------------------------------------------
__global__ __launch_bounds__(256) void finalize_kernel(
    const float* __restrict__ scores, const __bf16* __restrict__ ws_cs,
    const float* __restrict__ ws_ig, const float* __restrict__ ws_og,
    const float* __restrict__ pre,
    float* __restrict__ out_h, float* __restrict__ out_cso,
    float* __restrict__ out_probs) {
    const int b = blockIdx.x;
    const int t = threadIdx.x;
    __shared__ float s[CC];
    if (t < CC) s[t] = scores[b * CC + t];
    __syncthreads();
    float mx = s[0];
#pragma unroll
    for (int c = 1; c < CC; ++c) mx = fmaxf(mx, s[c]);
    float p[CC];
    float sum = 0.0f;
#pragma unroll
    for (int c = 0; c < CC; ++c) { p[c] = expf(s[c] - mx); sum += p[c]; }
    float inv = 1.0f / sum;
#pragma unroll
    for (int c = 0; c < CC; ++c) p[c] *= inv;
    if (t < CC) out_probs[(size_t)b * CC + t] = p[t];

    for (int h = t; h < HH; h += 256) {
        float att = 0.0f;
#pragma unroll
        for (int c = 0; c < CC; ++c)
            att += p[c] * (float)ws_cs[((size_t)c * BB + b) * HH + h];
        size_t idx = (size_t)b * HH + h;
        float ig = ws_ig[idx];
        float og = ws_og[idx];
        float pv = pre[idx];
        float cso = (1.0f - ig) * pv + ig * att;
        out_cso[idx] = cso;
        out_h[idx]   = og * tanhf(cso);
    }
}

// ---------------------------------------------------------------------------
// Launch
// ---------------------------------------------------------------------------
extern "C" void kernel_launch(void* const* d_in, const int* in_sizes, int n_in,
                              void* d_out, int out_size, void* d_ws, size_t ws_size,
                              hipStream_t stream) {
    const float* input_word   = (const float*)d_in[0];   // [B,I]
    const float* hidden       = (const float*)d_in[1];   // [B,H]
    const float* cell_pre     = (const float*)d_in[2];   // [B,H]
    const float* cell_x_W     = (const float*)d_in[3];   // [C,I,2H]
    const float* cell_x_b     = (const float*)d_in[4];   // [C,2H]
    const float* cell_h_W     = (const float*)d_in[5];   // [C,H,2H]
    const float* comp_x_W     = (const float*)d_in[6];   // [I,3H]
    const float* comp_x_b     = (const float*)d_in[7];   // [3H]
    const float* comp_h_W     = (const float*)d_in[8];   // [H,3H]
    const float* attn_in_W    = (const float*)d_in[9];   // [H,2H]
    const float* attn_cell_W  = (const float*)d_in[10];  // [H,2H]
    const float* atten_v      = (const float*)d_in[11];  // [2H]

    float* out = (float*)d_out;
    float* out_h     = out;                                       // [B,H]
    float* out_cso   = out + (size_t)BB * HH;                     // [B,H]
    float* out_cs    = out + (size_t)2 * BB * HH;                 // [B,C,H]
    float* out_probs = out + (size_t)2 * BB * HH + (size_t)BB * CC * HH; // [B,C]

    // Workspace carve (256B aligned)
    char* w = (char*)d_ws;
    auto carve = [&](size_t bytes) -> void* {
        void* p = (void*)w;
        w += (bytes + 255) & ~(size_t)255;
        return p;
    };
    __bf16* Xb     = (__bf16*)carve((size_t)BB * II * 2);             // bf16 input
    __bf16* Hb     = (__bf16*)carve((size_t)BB * HH * 2);             // bf16 hidden
    __bf16* WtCX   = (__bf16*)carve((size_t)CC * 2 * HH * II * 2);    // [C][2H][I]
    __bf16* WtCH   = (__bf16*)carve((size_t)CC * 2 * HH * HH * 2);    // [C][2H][H]
    __bf16* WtPX   = (__bf16*)carve((size_t)3 * HH * II * 2);         // [3H][I]
    __bf16* WtPH   = (__bf16*)carve((size_t)3 * HH * HH * 2);         // [3H][H]
    __bf16* WtAI   = (__bf16*)carve((size_t)2 * HH * HH * 2);         // [2H][H]
    __bf16* WtAC   = (__bf16*)carve((size_t)2 * HH * HH * 2);         // [2H][H]
    __bf16* ws_cs  = (__bf16*)carve((size_t)CC * BB * HH * 2);        // [C][B][H]
    __bf16* ws_cc  = (__bf16*)carve((size_t)BB * HH * 2);             // [B][H]
    float*  ws_u   = (float*)carve((size_t)BB * 2 * HH * 4);          // [B][2H]
    float*  ws_ig  = (float*)carve((size_t)BB * HH * 4);
    float*  ws_og  = (float*)carve((size_t)BB * HH * 4);
    float*  ws_sc  = (float*)carve((size_t)BB * CC * 4);

    // 1) Precision packing
    {
        size_t n = (size_t)BB * II;
        cvt_f32_bf16_kernel<<<(n + 255) / 256, 256, 0, stream>>>(input_word, Xb, n);
        n = (size_t)BB * HH;
        cvt_f32_bf16_kernel<<<(n + 255) / 256, 256, 0, stream>>>(hidden, Hb, n);
    }
    {
        dim3 tb(32, 8);
        transpose_pack_kernel<<<dim3(2 * HH / 32, II / 32, CC), tb, 0, stream>>>(cell_x_W, WtCX, II, 2 * HH);
        transpose_pack_kernel<<<dim3(2 * HH / 32, HH / 32, CC), tb, 0, stream>>>(cell_h_W, WtCH, HH, 2 * HH);
        transpose_pack_kernel<<<dim3(3 * HH / 32, II / 32, 1), tb, 0, stream>>>(comp_x_W, WtPX, II, 3 * HH);
        transpose_pack_kernel<<<dim3(3 * HH / 32, HH / 32, 1), tb, 0, stream>>>(comp_h_W, WtPH, HH, 3 * HH);
        transpose_pack_kernel<<<dim3(2 * HH / 32, HH / 32, 1), tb, 0, stream>>>(attn_in_W, WtAI, HH, 2 * HH);
        transpose_pack_kernel<<<dim3(2 * HH / 32, HH / 32, 1), tb, 0, stream>>>(attn_cell_W, WtAC, HH, 2 * HH);
    }
    zero_f32_kernel<<<(BB * CC + 255) / 256, 256, 0, stream>>>(ws_sc, (size_t)BB * CC);

    // 2) Cell gates (16 GEMMs, fused gate mix)
    cellgate_kernel<<<dim3(HH / 64, BB / 32, CC), 32, 0, stream>>>(
        Xb, Hb, WtCX, WtCH, cell_x_b, cell_pre, out_cs, ws_cs);

    // 3) Composition gates
    comp_kernel<<<dim3(3 * HH / 64, BB / 32), 32, 0, stream>>>(
        Xb, Hb, WtPX, WtPH, comp_x_b, ws_ig, ws_og, ws_cc);

    // 4) u = cc @ attn_in_W
    attn_u_kernel<<<dim3(2 * HH / 64, BB / 32), 32, 0, stream>>>(ws_cc, WtAI, ws_u);

    // 5) attention scores (split-N atomic accumulate)
    attn_score_kernel<<<dim3(2 * HH / 64, BB / 32, CC), 32, 0, stream>>>(
        ws_cs, WtAC, ws_u, atten_v, ws_sc);

    // 6) softmax + weighted sum + final gates
    finalize_kernel<<<BB, 256, 0, stream>>>(
        ws_sc, ws_cs, ws_ig, ws_og, cell_pre, out_h, out_cso, out_probs);
}
